// ClusterAttention_43602507989794
// MI455X (gfx1250) — matrix operand order repaired and checked by reading, not compiled, not run on hardware
//
#include <hip/hip_runtime.h>
#include <hip/hip_bf16.h>

// ---------------------------------------------------------------------------
// ClusterAttention for MI455X (gfx1250), wave32 + v_wmma_f32_16x16x32_bf16
// + global_load_async_to_lds_b128 (ASYNCcnt) for pure bf16 tile staging.
// Pipeline: [gathered QKV GEMM] -> [per-(z,h) attention] -> [proj GEMM]
// Intermediates in d_ws as bf16:  qkv_g (65536x1152) + attn_pre (65536x384)
// ---------------------------------------------------------------------------

typedef __attribute__((ext_vector_type(16))) __bf16 v16bf;
typedef __attribute__((ext_vector_type(8)))  float  v8f;
typedef unsigned int       u32;
typedef unsigned short     u16;
typedef unsigned long long u64;

#define B_   4
#define N_   16384
#define C_   384
#define H_   12
#define D_   2
#define Z_   1024
#define M_   64
#define CH_  32
#define QKVC (3 * C_)   // 1152

// padded LDS strides (u16 elems); multiples of 8 keep 16B alignment for b128 IO
#define AST  40   // A/B tile stride (32 data + 8 pad)
#define VST  72   // vt / Pb stride  (64 data + 8 pad)
#define SST  65   // Sb float stride (conflict-free column scans)

__device__ __forceinline__ u16 f2bf(float f) {
  union { float f; u32 u; } c; c.f = f;
  u32 r = c.u + 0x7FFFu + ((c.u >> 16) & 1u);   // round-to-nearest-even
  return (u16)(r >> 16);
}
__device__ __forceinline__ float bf2f(u16 h) {
  union { u32 u; float f; } c; c.u = ((u32)h) << 16; return c.f;
}

// ---- CDNA5 async global->LDS (tracked by ASYNCcnt, ISA ch.10 / 15.18.3) ----
// GV mode: 64-bit global address in VGPR pair; VDST = wave-relative LDS byte
// address (generic LDS pointer low 32 bits per ISA 10.2 aperture mapping).
__device__ __forceinline__ void async_copy16B(void* lds, const void* gsrc) {
  u32 ldsAddr = (u32)(u64)lds;
  asm volatile("global_load_async_to_lds_b128 %0, %1, off"
               :: "v"(ldsAddr), "v"((u64)gsrc) : "memory");
}
__device__ __forceinline__ void wait_async0() {
  asm volatile("s_wait_asynccnt 0" ::: "memory");
}

union V16 { uint4 q[2]; v16bf v; };

// A operand (16x32 bf16 tile): lds row-major [row][rowStride] u16.
// Lane L<16: row=L, elems0-7=K+0..7, elems8-15=K+16..23; L>=16: row=L-16, K+8.. / K+24..
__device__ __forceinline__ v16bf load_A(const u16* lds, int rowStride,
                                        int rowBase, int kBase, int lane) {
  int r = rowBase + (lane & 15);
  const u16* p = lds + r * rowStride + kBase + ((lane >> 4) << 3);
  V16 u;
  u.q[0] = *(const uint4*)(p);        // 8 bf16
  u.q[1] = *(const uint4*)(p + 16);   // 8 bf16, +32B
  return u.v;
}
// B operand (32x16 bf16 tile, K x N): lds stores column-contiguous-in-K: [n][rowStride].
// Lane L<16: col=L, elems0-7=K+0..7, elems8-15=K+16..23; L>=16: col=L-16, K+8.. / K+24..
__device__ __forceinline__ v16bf load_B(const u16* lds, int rowStride,
                                        int nBase, int kBase, int lane) {
  int n = nBase + (lane & 15);
  const u16* p = lds + n * rowStride + kBase + ((lane >> 4) << 3);
  V16 u;
  u.q[0] = *(const uint4*)(p);
  u.q[1] = *(const uint4*)(p + 16);
  return u.v;
}

// ---------------------------------------------------------------------------
// GEMM: out[row][col] = sum_k A[row][k] * W[k][col] + bias[col]
// GATHER=true : A rows are feat[batch_idx[row], member_idx[row], :] (fp32)
// GATHER=false: A is bf16 row-major [rows][KD] -> async global->LDS staging
// 256 threads / 8 waves; 128x64 tile; wave w owns rows w*16..+16, all 64 cols.
// Double-buffered LDS, fully unrolled K (KD/32 steps).
// ---------------------------------------------------------------------------
template <bool GATHER, int KD>
__global__ __launch_bounds__(256)
void gemm_bf16_wmma(const float* __restrict__ Af32, const u16* __restrict__ Abf,
                    const int* __restrict__ batch_idx, const int* __restrict__ member_idx,
                    const float* __restrict__ W, int ldw, int Ncols,
                    const float* __restrict__ bias,
                    u16* __restrict__ outBf, float* __restrict__ outF32) {
  constexpr int NIT = KD / 32;
  __shared__ u16 ldsA[2][128 * AST];   // [row][k]
  __shared__ u16 ldsB[2][64 * AST];    // [col][k]
  __shared__ u64 rowOff[128];

  const int tid  = threadIdx.x;
  const int lane = tid & 31;
  const int w    = tid >> 5;           // 0..7 : row sub-tile
  const int rowBase = blockIdx.y * 128;
  const int colBase = blockIdx.x * 64;

  if (tid < 128) {
    if (GATHER) {
      int zr = rowBase + tid;
      u64 bi = (u64)batch_idx[zr], mi = (u64)member_idx[zr];
      rowOff[tid] = (bi * N_ + mi) * (u64)C_;
    } else {
      rowOff[tid] = (u64)(rowBase + tid) * (u64)KD;
    }
  }
  __syncthreads();

  // staging thread mapping: A: 128 rows x 32 k, 16 elems/thread
  const int ar  = tid >> 1;            // 0..127
  const int ako = (tid & 1) * 16;      // 0 / 16
  // B: 32 k x 64 cols (transpose into [col][k]), 8 elems/thread
  const int bk  = tid >> 3;            // 0..31
  const int bc  = (tid & 7) * 8;       // 0..56

  auto stageA = [&](int buf, int k0) {
    if (GATHER) {
      const float4* s4 = (const float4*)(Af32 + rowOff[ar] + k0 + ako);
      float4 x0 = s4[0], x1 = s4[1], x2 = s4[2], x3 = s4[3];
      if (k0 + 32 < KD)                 // warm L2/WGP$ for next K-chunk
        __builtin_prefetch(Af32 + rowOff[ar] + k0 + 32 + ako, 0, 1);
      u16 t[16];
      t[0]=f2bf(x0.x); t[1]=f2bf(x0.y); t[2]=f2bf(x0.z); t[3]=f2bf(x0.w);
      t[4]=f2bf(x1.x); t[5]=f2bf(x1.y); t[6]=f2bf(x1.z); t[7]=f2bf(x1.w);
      t[8]=f2bf(x2.x); t[9]=f2bf(x2.y); t[10]=f2bf(x2.z); t[11]=f2bf(x2.w);
      t[12]=f2bf(x3.x); t[13]=f2bf(x3.y); t[14]=f2bf(x3.z); t[15]=f2bf(x3.w);
      *(uint4*)(&ldsA[buf][ar * AST + ako])     = *(uint4*)(t);
      *(uint4*)(&ldsA[buf][ar * AST + ako + 8]) = *(uint4*)(t + 8);
    } else {
      const u16* src = Abf + rowOff[ar] + k0 + ako;   // bf16, pure copy
      async_copy16B(&ldsA[buf][ar * AST + ako],     src);
      async_copy16B(&ldsA[buf][ar * AST + ako + 8], src + 8);
    }
  };
  auto stageB = [&](int buf, int k0) {
    const float4* s4 = (const float4*)(W + (u64)(k0 + bk) * ldw + colBase + bc);
    float4 x0 = s4[0], x1 = s4[1];
    if (k0 + 32 < KD)
      __builtin_prefetch(W + (u64)(k0 + 32 + bk) * ldw + colBase + bc, 0, 1);
    float xv[8] = {x0.x, x0.y, x0.z, x0.w, x1.x, x1.y, x1.z, x1.w};
#pragma unroll
    for (int i = 0; i < 8; i++) ldsB[buf][(bc + i) * AST + bk] = f2bf(xv[i]);
  };

  stageA(0, 0);
  stageB(0, 0);
  if (!GATHER) wait_async0();
  __syncthreads();

  v8f acc[4] = {v8f{}, v8f{}, v8f{}, v8f{}};

#pragma unroll
  for (int it = 0; it < NIT; ++it) {
    const int p = it & 1;
    if (it + 1 < NIT) {               // prefetch next tile into other buffer
      stageA(p ^ 1, (it + 1) * 32);
      stageB(p ^ 1, (it + 1) * 32);
    }
    v16bf a = load_A(ldsA[p], AST, w * 16, 0, lane);
#pragma unroll
    for (int ct = 0; ct < 4; ct++) {
      v16bf b = load_B(ldsB[p], AST, ct * 16, 0, lane);
      acc[ct] = __builtin_amdgcn_wmma_f32_16x16x32_bf16(false, a, false, b, (short)0, acc[ct], false, false);
    }
    if (!GATHER && it + 1 < NIT) wait_async0();
    __syncthreads();
  }

  // ---- store: C/D layout -> m = g + 8*(lane>=16), n = lane%16 ----
  const int mL = w * 16 + ((lane >> 4) << 3);
  const int nL = lane & 15;
#pragma unroll
  for (int ct = 0; ct < 4; ct++) {
#pragma unroll
    for (int g = 0; g < 8; g++) {
      int m = rowBase + mL + g;
      int c = colBase + ct * 16 + nL;
      float v = acc[ct][g] + bias[c];
      if (outBf) outBf[(u64)m * Ncols + c] = f2bf(v);
      else       outF32[(u64)m * Ncols + c] = v;
    }
  }
}

// ---------------------------------------------------------------------------
// Attention per (z,h): S = scale*q*k^T + 2*cf*cf^T - nrm_m - nrm_k
//                        + (pp[k]-pp[m]+b_pos[h]) + (1-mask[k])*(-100)
// softmax rows, O = P*V.  128 threads / 4 waves; wave w owns rows w*16..+16.
// ---------------------------------------------------------------------------
__global__ __launch_bounds__(128)
void attn_kernel(const u16* __restrict__ qkv,            // [Z*M][1152] bf16
                 const float* __restrict__ cluster_feat, // (B,N,CH)
                 const float* __restrict__ pos,          // (B,N,D)
                 const float* __restrict__ mask,         // (B,N,1)
                 const int* __restrict__ batch_idx, const int* __restrict__ member_idx,
                 const float* __restrict__ w_pos,        // (D,H)
                 const float* __restrict__ b_pos,        // (H)
                 u16* __restrict__ outPre) {              // [Z*M][384] bf16
  __shared__ u16 qs[64 * AST];    // q * scale
  __shared__ u16 ks[64 * AST];    // k  ([m'][ch] == B layout for q*k^T)
  __shared__ u16 cfs[64 * AST];   // cf * sqrt2 (A and B roles)
  __shared__ u16 vt[32 * VST];    // V^T: [ch][m'] == B layout for P*V
  __shared__ float Sb[64 * SST];
  __shared__ u16 Pb[64 * VST];
  __shared__ float rowAdd[64], colAdd[64];

  const int h = blockIdx.x;
  const int z = blockIdx.y;
  const int tid  = threadIdx.x;
  const int lane = tid & 31;
  const int w    = tid >> 5;                 // 0..3
  const float scale = 0.17677669529663687f;  // 32^-0.5
  const float sqrt2 = 1.41421356237309515f;

  // ---- stage q,k,v,cf (each thread: half a row) ----
  {
    int m = tid >> 1, half = tid & 1;
    const u16* base = qkv + (u64)(z * 64 + m) * QKVC + h * CH_ + half * 16;
    // k: pure bf16 copy -> async global->LDS (ASYNCcnt path)
    async_copy16B(&ks[m * AST + half * 16],     base + C_);
    async_copy16B(&ks[m * AST + half * 16 + 8], base + C_ + 8);
#pragma unroll
    for (int i = 0; i < 16; i++) {
      qs[m * AST + half * 16 + i] = f2bf(bf2f(base[i]) * scale);
      vt[(half * 16 + i) * VST + m] = base[2 * C_ + i];
    }
    int bi = batch_idx[z * 64 + m], mi = member_idx[z * 64 + m];
    const float4* cf4 = (const float4*)(cluster_feat + ((u64)bi * N_ + mi) * CH_ + half * 16);
#pragma unroll
    for (int j = 0; j < 4; j++) {
      float4 x = cf4[j];
      float xv[4] = {x.x, x.y, x.z, x.w};
#pragma unroll
      for (int i = 0; i < 4; i++)
        cfs[m * AST + half * 16 + j * 4 + i] = f2bf(xv[i] * sqrt2);
    }
  }
  wait_async0();
  __syncthreads();

  // ---- rank-1 bias terms ----
  if (tid < 64) {
    int m = tid;
    float nrm = 0.f;
#pragma unroll
    for (int i = 0; i < 32; i++) { float f = bf2f(cfs[m * AST + i]); nrm += f * f; }
    nrm *= 0.5f;  // undo sqrt2^2
    int bi = batch_idx[z * 64 + m], mi = member_idx[z * 64 + m];
    const float* pr = pos + ((u64)bi * N_ + mi) * D_;
    float pp = pr[0] * w_pos[0 * H_ + h] + pr[1] * w_pos[1 * H_ + h];
    float mk = mask[(u64)bi * N_ + mi];
    colAdd[m] = -nrm + pp + (1.0f - mk) * (-100.0f);
    rowAdd[m] = -nrm - pp + b_pos[h];
  }
  __syncthreads();

  // ---- S = 2*cf*cf^T + scale*q*k^T (chained WMMAs into one accumulator) ----
  {
    v16bf a_cf = load_A(cfs, AST, w * 16, 0, lane);
    v16bf a_q  = load_A(qs,  AST, w * 16, 0, lane);
#pragma unroll
    for (int ct = 0; ct < 4; ct++) {
      v16bf b_cf = load_B(cfs, AST, ct * 16, 0, lane);
      v16bf b_k  = load_B(ks,  AST, ct * 16, 0, lane);
      v8f acc = {};
      acc = __builtin_amdgcn_wmma_f32_16x16x32_bf16(false, a_cf, false, b_cf, (short)0, acc, false, false);
      acc = __builtin_amdgcn_wmma_f32_16x16x32_bf16(false, a_q,  false, b_k,  (short)0, acc, false, false);
      int mL = w * 16 + ((lane >> 4) << 3);
      int n  = ct * 16 + (lane & 15);
#pragma unroll
      for (int g = 0; g < 8; g++)
        Sb[(mL + g) * SST + n] = acc[g] + rowAdd[mL + g] + colAdd[n];
    }
  }
  __syncthreads();

  // ---- row softmax -> Pb (bf16) ----
  if (tid < 64) {
    float mx = -3.0e38f;
#pragma unroll 4
    for (int i = 0; i < 64; i++) mx = fmaxf(mx, Sb[tid * SST + i]);
    float sum = 0.f;
#pragma unroll 4
    for (int i = 0; i < 64; i++) { float e = __expf(Sb[tid * SST + i] - mx); Sb[tid * SST + i] = e; sum += e; }
    float inv = 1.0f / sum;
#pragma unroll 4
    for (int i = 0; i < 64; i++) Pb[tid * VST + i] = f2bf(Sb[tid * SST + i] * inv);
  }
  __syncthreads();

  // ---- O = P * V : K=64 (two 32-chunks), N=32 (two 16-col tiles) ----
  {
    v8f o0 = {}, o1 = {};
#pragma unroll
    for (int kc = 0; kc < 2; kc++) {
      v16bf a  = load_A(Pb, VST, w * 16, kc * 32, lane);
      v16bf b0 = load_B(vt, VST, 0,  kc * 32, lane);
      v16bf b1 = load_B(vt, VST, 16, kc * 32, lane);
      o0 = __builtin_amdgcn_wmma_f32_16x16x32_bf16(false, a, false, b0, (short)0, o0, false, false);
      o1 = __builtin_amdgcn_wmma_f32_16x16x32_bf16(false, a, false, b1, (short)0, o1, false, false);
    }
    int mL = w * 16 + ((lane >> 4) << 3);
    int n  = lane & 15;
#pragma unroll
    for (int g = 0; g < 8; g++) {
      u64 row = (u64)(z * 64 + mL + g);
      outPre[row * C_ + h * CH_ + n]      = f2bf(o0[g]);
      outPre[row * C_ + h * CH_ + 16 + n] = f2bf(o1[g]);
    }
  }
}

// ---------------------------------------------------------------------------
extern "C" void kernel_launch(void* const* d_in, const int* in_sizes, int n_in,
                              void* d_out, int out_size, void* d_ws, size_t ws_size,
                              hipStream_t stream) {
  const float* pos          = (const float*)d_in[0];
  const float* feat         = (const float*)d_in[1];
  const float* cluster_feat = (const float*)d_in[2];
  const float* mask         = (const float*)d_in[3];
  const int*   member_idx   = (const int*)d_in[4];
  const int*   batch_idx    = (const int*)d_in[5];
  // d_in[6..8]: k / valid_row_idx / attend_means scalars (fixed path, ignored)
  const float* w_qkv  = (const float*)d_in[9];
  const float* b_qkv  = (const float*)d_in[10];
  const float* w_pos  = (const float*)d_in[11];
  const float* b_pos  = (const float*)d_in[12];
  const float* w_proj = (const float*)d_in[13];
  const float* b_proj = (const float*)d_in[14];
  float* out = (float*)d_out;

  // workspace: bf16 qkv_g (151 MB) + bf16 attn_pre (50 MB)
  u16* qkv_g    = (u16*)d_ws;
  u16* attn_pre = qkv_g + (u64)Z_ * M_ * QKVC;

  // 1) gathered QKV GEMM: (65536 x 384) x (384 x 1152)
  dim3 g1(QKVC / 64, (Z_ * M_) / 128);  // 18 x 512
  hipLaunchKernelGGL((gemm_bf16_wmma<true, C_>), g1, dim3(256), 0, stream,
                     feat, (const u16*)nullptr, batch_idx, member_idx,
                     w_qkv, QKVC, QKVC, b_qkv, qkv_g, (float*)nullptr);

  // 2) attention per (z,h)
  dim3 g2(H_, Z_);
  hipLaunchKernelGGL(attn_kernel, g2, dim3(128), 0, stream,
                     qkv_g, cluster_feat, pos, mask, batch_idx, member_idx,
                     w_pos, b_pos, attn_pre);

  // 3) projection GEMM: (65536 x 384) x (384 x 384) -> fp32 out
  dim3 g3(C_ / 64, (Z_ * M_) / 128);    // 6 x 512
  hipLaunchKernelGGL((gemm_bf16_wmma<false, C_>), g3, dim3(256), 0, stream,
                     (const float*)nullptr, attn_pre, batch_idx, member_idx,
                     w_proj, C_, C_, b_proj, (u16*)nullptr, out);
}
